// MultiHeadAttention_17119739641971
// MI455X (gfx1250) — matrix operand order repaired and checked by reading, not compiled, not run on hardware
//
#include <hip/hip_runtime.h>
#include <hip/hip_bf16.h>

// ---------------------------------------------------------------------------
// MHA forward for MI455X (gfx1250, wave32, WMMA).
// B=4, S=1024, D=1024, H=16, DK=64.
// All GEMMs via v_wmma_f32_16x16x32_f16 (f16 A/B, f32 accumulate).
// V tiles staged to LDS with gfx1250 async copies (ASYNCcnt, double-buffered).
// ---------------------------------------------------------------------------

#define B_  4
#define S_  1024
#define D_  1024
#define H_  16
#define DK_ 64
#define BH_ (B_ * H_)
#define M_  (B_ * S_)   // 4096 rows in the projection GEMMs

typedef __attribute__((ext_vector_type(16))) _Float16 v16h;
typedef __attribute__((ext_vector_type(8)))  float    v8f;

#define WMMA_F16(a, b, c) \
  __builtin_amdgcn_wmma_f32_16x16x32_f16(false, (a), false, (b), (short)0, (c), false, false)

// ---------------------------------------------------------------------------
// Fragment loaders (wave32), contiguous k order used for both A and B (not the ISA table order)
// A 16x32 f16 (MxK): lane l -> M = l&15, half = l>>4;
//   vgpr i holds K = half*16 + 2*i  (+0/+1 in the dword)
// ---------------------------------------------------------------------------
__device__ inline v16h load_a_frag(const _Float16* base, int ld, int lane) {
  const int m = lane & 15, hl = lane >> 4;
  const _Float16* row = base + (size_t)m * ld;
  v16h a;
#pragma unroll
  for (int i = 0; i < 8; ++i) {
    const int k = (hl << 4) + 2 * i;
    a[2 * i]     = row[k];
    a[2 * i + 1] = row[k + 1];
  }
  return a;
}

// B 32x16 f16 (KxN) where memory holds the (N,K) matrix row-major
// (i.e. B[k][n] = src[n*ld + k]; used for W^T and K^T operands).
// lane l -> N = l&15, half = l>>4; vgpr i holds K = half*16 + 2*i (+0/+1).
__device__ inline v16h load_b_frag_nk(const _Float16* src, int ld, int lane) {
  const int n = lane & 15, hl = lane >> 4;
  const _Float16* row = src + (size_t)n * ld + (hl << 4);
  v16h b;
#pragma unroll
  for (int i = 0; i < 8; ++i) {
    b[2 * i]     = row[2 * i];
    b[2 * i + 1] = row[2 * i + 1];
  }
  return b;
}

// B 32x16 f16 (KxN) where memory holds (K,N) row-major (used for the V operand).
__device__ inline v16h load_b_frag_kn(const _Float16* src, int ld, int lane) {
  const int n = lane & 15, hl = lane >> 4;
  v16h b;
#pragma unroll
  for (int i = 0; i < 8; ++i) {
    const int k = (hl << 4) + 2 * i;
    b[2 * i]     = src[(size_t)k * ld + n];
    b[2 * i + 1] = src[(size_t)(k + 1) * ld + n];
  }
  return b;
}

// ---------------------------------------------------------------------------
// fp32 -> f16 staging
// ---------------------------------------------------------------------------
__global__ void cvt_f32_to_f16_kernel(const float* __restrict__ src,
                                      _Float16* __restrict__ dst, int n) {
  int i = blockIdx.x * blockDim.x + threadIdx.x;
  if (i < n) dst[i] = (_Float16)src[i];
}

// ---------------------------------------------------------------------------
// Fused: Y = relu( (X * W^T + b) [head h cols] * Wf^T + bf ), head-split store.
// One wave per (16-row block, head): 128 + 8 WMMAs.
// ---------------------------------------------------------------------------
__global__ __launch_bounds__(32)
void proj_ff_kernel(const _Float16* __restrict__ X16,   // (M_, D_)
                    const _Float16* __restrict__ W16,   // (D_, D_) as (out,in)
                    const float*    __restrict__ bias,  // (D_)
                    const _Float16* __restrict__ Wf16,  // (DK_, DK_)
                    const float*    __restrict__ bf,    // (DK_)
                    _Float16*       __restrict__ Yh) {  // (BH_, S_, DK_)
  __shared__ _Float16 lds[16 * 72];

  const int lane = threadIdx.x;
  const int m0   = blockIdx.x * 16;   // row block in M_
  const int h    = blockIdx.y;        // head
  const int nl   = lane & 15, hl = lane >> 4;

  v8f acc[4] = {};
  const _Float16* Xrow = X16 + (size_t)m0 * D_;
  for (int kc = 0; kc < D_; kc += 32) {
    __builtin_prefetch(Xrow + kc + 64, 0, 1);
    v16h a = load_a_frag(Xrow + kc, D_, lane);
#pragma unroll
    for (int t = 0; t < 4; ++t) {
      v16h b = load_b_frag_nk(W16 + (size_t)(h * DK_ + t * 16) * D_ + kc, D_, lane);
      acc[t] = WMMA_F16(a, b, acc[t]);
    }
  }

  // bias add, spill 16x64 f16 tile to LDS for fragment relayout
#pragma unroll
  for (int t = 0; t < 4; ++t) {
    const float bv = bias[h * DK_ + t * 16 + nl];
#pragma unroll
    for (int r = 0; r < 8; ++r) {
      const int mm = hl * 8 + r;
      lds[mm * 72 + t * 16 + nl] = (_Float16)(acc[t][r] + bv);
    }
  }
  __syncthreads();

  // per-head feed-forward GEMM (16x64 * 64x64)
  v8f acc2[4] = {};
#pragma unroll
  for (int kc = 0; kc < 2; ++kc) {
    v16h a = load_a_frag(lds + kc * 32, 72, lane);
#pragma unroll
    for (int t = 0; t < 4; ++t) {
      v16h b = load_b_frag_nk(Wf16 + (size_t)(t * 16) * DK_ + kc * 32, DK_, lane);
      acc2[t] = WMMA_F16(a, b, acc2[t]);
    }
  }

  // ReLU + head-split f16 store: (bh, s, dk)
#pragma unroll
  for (int t = 0; t < 4; ++t) {
    const float bv = bf[t * 16 + nl];
#pragma unroll
    for (int r = 0; r < 8; ++r) {
      const int mg = m0 + hl * 8 + r;
      const int bb = mg >> 10, s = mg & (S_ - 1);
      float v = acc2[t][r] + bv;
      v = fmaxf(v, 0.0f);
      Yh[((size_t)(bb * H_ + h) * S_ + s) * DK_ + t * 16 + nl] = (_Float16)v;
    }
  }
}

// ---------------------------------------------------------------------------
// Flash-style attention + fused out-projection.
// One wave per (16-query block, head). Online softmax; no SxS tensor.
// V KV-blocks (32x64 f16 = 4KB, contiguous) are staged into LDS with
// gfx1250 async global->LDS copies, double-buffered across the KV loop.
// ---------------------------------------------------------------------------
__device__ inline void stage_v_async(const _Float16* __restrict__ gsrc,
                                     _Float16* lds_dst, int lane) {
  // 32 rows x 64 f16 = 4096 B contiguous; 8 wave-wide 128b async copies.
  const unsigned d0 = (unsigned)(size_t)lds_dst + (unsigned)lane * 16u;
#pragma unroll
  for (int i = 0; i < 8; ++i) {
    const _Float16* g = gsrc + i * 256 + lane * 8;  // 16 B per lane
    const unsigned  d = d0 + (unsigned)i * 512u;
    asm volatile("global_load_async_to_lds_b128 %0, %1, off"
                 :: "v"(d), "v"(g)
                 : "memory");
  }
}

__global__ __launch_bounds__(32)
void attn_kernel(const _Float16* __restrict__ Qh,
                 const _Float16* __restrict__ Kh,
                 const _Float16* __restrict__ Vh,
                 const int*      __restrict__ mask,   // (BH_, S_)
                 const _Float16* __restrict__ Wo16,   // (DK_, DK_)
                 const float*    __restrict__ bo,     // (DK_)
                 float*          __restrict__ out) {  // (B_, S_, D_)
  __shared__ _Float16 ldsP[16 * 40];      // 16x32 P tile (+pad)
  __shared__ _Float16 ldsO[16 * 72];      // 16x64 O tile (+pad)
  __shared__ _Float16 ldsV[2][32 * DK_];  // double-buffered V block (4KB each)

  const int lane = threadIdx.x;
  const int qb   = blockIdx.x;    // query block (0..63)
  const int bh   = blockIdx.y;    // batch*head (0..63)
  const int nl   = lane & 15, hl = lane >> 4;

  const _Float16* Qbase = Qh + ((size_t)bh * S_ + (size_t)qb * 16) * DK_;
  const _Float16* Kbase = Kh + (size_t)bh * S_ * DK_;
  const _Float16* Vbase = Vh + (size_t)bh * S_ * DK_;

  v16h qa[2];
  qa[0] = load_a_frag(Qbase, DK_, lane);
  qa[1] = load_a_frag(Qbase + 32, DK_, lane);

  float mrow[8], lrow[8];
#pragma unroll
  for (int r = 0; r < 8; ++r) { mrow[r] = -3.0e38f; lrow[r] = 0.0f; }
  v8f o[4] = {};

  // prologue: stage V block 0
  stage_v_async(Vbase, &ldsV[0][0], lane);

  for (int j = 0; j < S_; j += 32) {
    const int bj = j >> 5;
    __builtin_prefetch(Kbase + (size_t)(j + 32) * DK_, 0, 1);

    // S = Q * K^T  (16x32 keys)
    v8f sc[2] = {};
#pragma unroll
    for (int nt = 0; nt < 2; ++nt) {
#pragma unroll
      for (int kc = 0; kc < 2; ++kc) {
        v16h b = load_b_frag_nk(Kbase + (size_t)(j + nt * 16) * DK_ + kc * 32, DK_, lane);
        sc[nt] = WMMA_F16(qa[kc], b, sc[nt]);
      }
    }

    // scale 1/sqrt(DK) and padding mask (mask==0 -> -10000)
    const float mk0 = (mask[bh * S_ + j + nl]      == 0) ? -10000.0f : 0.0f;
    const float mk1 = (mask[bh * S_ + j + 16 + nl] == 0) ? -10000.0f : 0.0f;
#pragma unroll
    for (int r = 0; r < 8; ++r) {
      sc[0][r] = sc[0][r] * 0.125f + mk0;
      sc[1][r] = sc[1][r] * 0.125f + mk1;
    }

    // online softmax: row max / exp / row sum (rows live in 16-lane halves)
#pragma unroll
    for (int r = 0; r < 8; ++r) {
      float v = fmaxf(sc[0][r], sc[1][r]);
      v = fmaxf(v, __shfl_xor(v, 1, 32));
      v = fmaxf(v, __shfl_xor(v, 2, 32));
      v = fmaxf(v, __shfl_xor(v, 4, 32));
      v = fmaxf(v, __shfl_xor(v, 8, 32));
      const float mnew = fmaxf(mrow[r], v);
      const float p0 = __expf(sc[0][r] - mnew);
      const float p1 = __expf(sc[1][r] - mnew);
      float rs = p0 + p1;
      rs += __shfl_xor(rs, 1, 32);
      rs += __shfl_xor(rs, 2, 32);
      rs += __shfl_xor(rs, 4, 32);
      rs += __shfl_xor(rs, 8, 32);
      const float c = __expf(mrow[r] - mnew);
      lrow[r] = lrow[r] * c + rs;
      mrow[r] = mnew;
#pragma unroll
      for (int t = 0; t < 4; ++t) o[t][r] *= c;
      const int mm = hl * 8 + r;
      ldsP[mm * 40 + nl]      = (_Float16)p0;
      ldsP[mm * 40 + 16 + nl] = (_Float16)p1;
    }

    // stage next V block while this block's softmax/PV proceed
    if (j + 32 < S_) {
      stage_v_async(Vbase + (size_t)(j + 32) * DK_, &ldsV[(bj + 1) & 1][0], lane);
      asm volatile("s_wait_asynccnt 0x8" ::: "memory");  // oldest 8 (block bj) done
    } else {
      asm volatile("s_wait_asynccnt 0x0" ::: "memory");
    }
    __syncthreads();

    // O += P * V   (16x32 * 32x64), V from LDS
    v16h pa = load_a_frag(ldsP, 40, lane);
    const _Float16* Vb = &ldsV[bj & 1][0];
#pragma unroll
    for (int t = 0; t < 4; ++t) {
      v16h vb = load_b_frag_kn(Vb + t * 16, DK_, lane);
      o[t] = WMMA_F16(pa, vb, o[t]);
    }
    __syncthreads();
  }

  // normalize rows, spill to LDS for the fused out-projection
#pragma unroll
  for (int t = 0; t < 4; ++t) {
#pragma unroll
    for (int r = 0; r < 8; ++r) {
      const int mm = hl * 8 + r;
      ldsO[mm * 72 + t * 16 + nl] = (_Float16)(o[t][r] / lrow[r]);
    }
  }
  __syncthreads();

  // out = O * Wo^T + bo, write fp32 in (B, S, H*DK) layout
  v8f acc2[4] = {};
#pragma unroll
  for (int kc = 0; kc < 2; ++kc) {
    v16h a = load_a_frag(ldsO + kc * 32, 72, lane);
#pragma unroll
    for (int t = 0; t < 4; ++t) {
      v16h b = load_b_frag_nk(Wo16 + (size_t)(t * 16) * DK_ + kc * 32, DK_, lane);
      acc2[t] = WMMA_F16(a, b, acc2[t]);
    }
  }
  const int bb = bh >> 4, h = bh & 15;
#pragma unroll
  for (int t = 0; t < 4; ++t) {
    const float bv = bo[t * 16 + nl];
#pragma unroll
    for (int r = 0; r < 8; ++r) {
      const int s = qb * 16 + hl * 8 + r;
      out[((size_t)(bb * S_ + s)) * D_ + h * DK_ + t * 16 + nl] = acc2[t][r] + bv;
    }
  }
}

// ---------------------------------------------------------------------------
// Host launcher
// ---------------------------------------------------------------------------
extern "C" void kernel_launch(void* const* d_in, const int* in_sizes, int n_in,
                              void* d_out, int out_size, void* d_ws, size_t ws_size,
                              hipStream_t stream) {
  (void)in_sizes; (void)n_in; (void)out_size; (void)ws_size;

  const float* q    = (const float*)d_in[0];
  const float* k    = (const float*)d_in[1];
  const float* v    = (const float*)d_in[2];
  const int*   mask = (const int*)  d_in[3];
  const float* Wq   = (const float*)d_in[4];
  const float* bq   = (const float*)d_in[5];
  const float* Wk   = (const float*)d_in[6];
  const float* bk   = (const float*)d_in[7];
  const float* Wv   = (const float*)d_in[8];
  const float* bv   = (const float*)d_in[9];
  const float* Wqf  = (const float*)d_in[10];
  const float* bqf  = (const float*)d_in[11];
  const float* Wkf  = (const float*)d_in[12];
  const float* bkf  = (const float*)d_in[13];
  const float* Wvf  = (const float*)d_in[14];
  const float* bvf  = (const float*)d_in[15];
  const float* Wo   = (const float*)d_in[16];
  const float* bo   = (const float*)d_in[17];
  float* out = (float*)d_out;

  // workspace layout (f16 elements)
  _Float16* ws = (_Float16*)d_ws;
  const size_t NXD = (size_t)M_ * D_;         // 4,194,304
  const size_t NWW = (size_t)D_ * D_;         // 1,048,576
  const size_t NWF = (size_t)DK_ * DK_;       // 4,096
  const size_t NHD = (size_t)BH_ * S_ * DK_;  // 4,194,304

  _Float16* q16  = ws;                  // + NXD
  _Float16* k16  = q16  + NXD;
  _Float16* v16  = k16  + NXD;
  _Float16* Wq16 = v16  + NXD;
  _Float16* Wk16 = Wq16 + NWW;
  _Float16* Wv16 = Wk16 + NWW;
  _Float16* Wqf16 = Wv16  + NWW;
  _Float16* Wkf16 = Wqf16 + NWF;
  _Float16* Wvf16 = Wkf16 + NWF;
  _Float16* Wo16  = Wvf16 + NWF;
  _Float16* Qh = Wo16 + NWF;
  _Float16* Kh = Qh + NHD;
  _Float16* Vh = Kh + NHD;

  // stage fp32 -> f16
  const int CT = 256;
  cvt_f32_to_f16_kernel<<<(int)((NXD + CT - 1) / CT), CT, 0, stream>>>(q, q16, (int)NXD);
  cvt_f32_to_f16_kernel<<<(int)((NXD + CT - 1) / CT), CT, 0, stream>>>(k, k16, (int)NXD);
  cvt_f32_to_f16_kernel<<<(int)((NXD + CT - 1) / CT), CT, 0, stream>>>(v, v16, (int)NXD);
  cvt_f32_to_f16_kernel<<<(int)((NWW + CT - 1) / CT), CT, 0, stream>>>(Wq, Wq16, (int)NWW);
  cvt_f32_to_f16_kernel<<<(int)((NWW + CT - 1) / CT), CT, 0, stream>>>(Wk, Wk16, (int)NWW);
  cvt_f32_to_f16_kernel<<<(int)((NWW + CT - 1) / CT), CT, 0, stream>>>(Wv, Wv16, (int)NWW);
  cvt_f32_to_f16_kernel<<<(int)((NWF + CT - 1) / CT), CT, 0, stream>>>(Wqf, Wqf16, (int)NWF);
  cvt_f32_to_f16_kernel<<<(int)((NWF + CT - 1) / CT), CT, 0, stream>>>(Wkf, Wkf16, (int)NWF);
  cvt_f32_to_f16_kernel<<<(int)((NWF + CT - 1) / CT), CT, 0, stream>>>(Wvf, Wvf16, (int)NWF);
  cvt_f32_to_f16_kernel<<<(int)((NWF + CT - 1) / CT), CT, 0, stream>>>(Wo, Wo16, (int)NWF);

  // projections + per-head FF + ReLU, head-split f16
  dim3 pgrid(M_ / 16, H_);
  proj_ff_kernel<<<pgrid, 32, 0, stream>>>(q16, Wq16, bq, Wqf16, bqf, Qh);
  proj_ff_kernel<<<pgrid, 32, 0, stream>>>(k16, Wk16, bk, Wkf16, bkf, Kh);
  proj_ff_kernel<<<pgrid, 32, 0, stream>>>(v16, Wv16, bv, Wvf16, bvf, Vh);

  // flash attention + fused out-projection
  dim3 agrid(S_ / 16, BH_);
  attn_kernel<<<agrid, 32, 0, stream>>>(Qh, Kh, Vh, mask, Wo16, bo, out);
}
